// RelationalGATLayer_24034636989286
// MI455X (gfx1250) — compile-verified, matrix operand order checked by reading, not run on hardware
//
#include <hip/hip_runtime.h>

#define NND 50000
#define NED 800000
#define HD 128
#define NH 4
#define NR 4
#define DH 32
#define NEG_SLOPE 0.2f
#define LN_EPS 1e-5f

typedef __attribute__((ext_vector_type(16))) __bf16 v16bf;
typedef __attribute__((ext_vector_type(8)))  float  v8f;

union Frag16 { v16bf v; unsigned u[8]; };

__device__ __forceinline__ unsigned short f2bf(float f) {
    unsigned u = __float_as_uint(f);
    unsigned r = (u + 0x7FFFu + ((u >> 16) & 1u)) >> 16;   // round-to-nearest-even
    return (unsigned short)r;
}
__device__ __forceinline__ unsigned pack2bf(float a, float b) {
    return (unsigned)f2bf(a) | ((unsigned)f2bf(b) << 16);
}
__device__ __forceinline__ void atomAddF32(float* p, float v) {
    __hip_atomic_fetch_add(p, v, __ATOMIC_RELAXED, __HIP_MEMORY_SCOPE_AGENT);
}

// ---- convert x (f32) -> bf16 pairs -----------------------------------------
__global__ void cvt_x_kernel(const float* __restrict__ x, unsigned* __restrict__ xb, int npairs) {
    int t = blockIdx.x * blockDim.x + threadIdx.x;
    if (t >= npairs) return;
    xb[t] = pack2bf(x[2 * t], x[2 * t + 1]);
}

// ---- convert W [R,K,N] f32 -> Wt [R,N,K] bf16 pairs ------------------------
__global__ void cvt_w_kernel(const float* __restrict__ W, unsigned* __restrict__ wt) {
    int t = blockIdx.x * blockDim.x + threadIdx.x;      // R*128*64
    if (t >= NR * HD * 64) return;
    int kp = t & 63;
    int n  = (t >> 6) & 127;
    int r  = t >> 13;
    const float* Wr = W + (size_t)r * HD * HD;
    wt[((size_t)(r * HD + n) * 64) + kp] = pack2bf(Wr[(2 * kp) * HD + n], Wr[(2 * kp + 1) * HD + n]);
}

// ---- H[r] = x @ W[r], bf16 WMMA, one wave -> 16 rows x 128 cols ------------
__global__ void gemm_kernel(const unsigned* __restrict__ xb, const unsigned* __restrict__ wt,
                            float* __restrict__ H) {
    int r    = blockIdx.y;
    int wave = threadIdx.x >> 5;
    int lane = threadIdx.x & 31;
    int tileM = blockIdx.x * 128 + wave * 16;
    if (tileM >= NND) return;                 // NND % 16 == 0 -> whole-wave uniform
    int m  = lane & 15;
    int hi = lane >> 4;                       // K offset +8 for upper half-wave
    const unsigned* arow  = xb + (size_t)(tileM + m) * 64 + hi * 4;
    const unsigned* wbase = wt + (size_t)(r * HD + m) * 64 + hi * 4;

    v8f acc[8] = {};
    #pragma unroll
    for (int ks = 0; ks < 4; ++ks) {
        int kb = ks * 16;                      // dword offset of K-block
        Frag16 a;
        #pragma unroll
        for (int j = 0; j < 4; ++j) { a.u[j] = arow[kb + j]; a.u[4 + j] = arow[kb + 8 + j]; }
        #pragma unroll
        for (int nt = 0; nt < 8; ++nt) {
            const unsigned* bp = wbase + (size_t)nt * 16 * 64;
            Frag16 b;
            #pragma unroll
            for (int j = 0; j < 4; ++j) { b.u[j] = bp[kb + j]; b.u[4 + j] = bp[kb + 8 + j]; }
            acc[nt] = __builtin_amdgcn_wmma_f32_16x16x32_bf16(
                false, a.v, false, b.v, (short)0, acc[nt], false, false);
        }
    }
    float* Hr = H + ((size_t)r * NND + tileM) * HD;
    #pragma unroll
    for (int nt = 0; nt < 8; ++nt)
        #pragma unroll
        for (int j = 0; j < 8; ++j)
            Hr[(size_t)(j + hi * 8) * HD + nt * 16 + m] = acc[nt][j];
}

// ---- per-node logit halves: s[r][n][h], t[r][n][h] -------------------------
__global__ void node_attn_kernel(const float* __restrict__ H,
                                 const float* __restrict__ asrc, const float* __restrict__ adst,
                                 float* __restrict__ sval, float* __restrict__ tval) {
    int t = blockIdx.x * blockDim.x + threadIdx.x;     // N*R*H
    if (t >= NND * NR * NH) return;
    int h = t & 3, r = (t >> 2) & 3, n = t >> 4;
    const float* hp = H + ((size_t)r * NND + n) * HD + h * DH;
    const float* as = asrc + (size_t)(r * NH + h) * DH;
    const float* ad = adst + (size_t)(r * NH + h) * DH;
    float s = 0.f, tt = 0.f;
    #pragma unroll
    for (int d = 0; d < DH; ++d) { float hv = hp[d]; s += hv * as[d]; tt += hv * ad[d]; }
    sval[((size_t)r * NND + n) * NH + h] = s;
    tval[((size_t)r * NND + n) * NH + h] = tt;
}

// ---- pass 1: per-dst running max (buffer starts at 0, matching reference) --
__global__ void edge_max_kernel(const int* __restrict__ ei, const int* __restrict__ et,
                                const float* __restrict__ sval, const float* __restrict__ tval,
                                int* __restrict__ emax) {
    int e = blockIdx.x * blockDim.x + threadIdx.x;
    if (e >= NED) return;
    int s = ei[e], d = ei[NED + e], r = et[e];
    const float* sp = sval + ((size_t)r * NND + s) * NH;
    const float* tp = tval + ((size_t)r * NND + d) * NH;
    int* mp = emax + ((size_t)r * NND + d) * NH;
    #pragma unroll
    for (int h = 0; h < NH; ++h) {
        float v = sp[h] + tp[h];
        v = v > 0.f ? v : NEG_SLOPE * v;
        if (v > 0.f) atomicMax(mp + h, __float_as_int(v));  // nonneg floats compare as ints
    }
}

// ---- pass 2: exp(e - max), store per-edge, atomic sum per dst --------------
__global__ void edge_sum_kernel(const int* __restrict__ ei, const int* __restrict__ et,
                                const float* __restrict__ sval, const float* __restrict__ tval,
                                const int* __restrict__ emax, float* __restrict__ esum,
                                float* __restrict__ eexp) {
    int e = blockIdx.x * blockDim.x + threadIdx.x;
    if (e >= NED) return;
    int s = ei[e], d = ei[NED + e], r = et[e];
    const float* sp = sval + ((size_t)r * NND + s) * NH;
    const float* tp = tval + ((size_t)r * NND + d) * NH;
    const int*   mp = emax + ((size_t)r * NND + d) * NH;
    float*       qp = esum + ((size_t)r * NND + d) * NH;
    #pragma unroll
    for (int h = 0; h < NH; ++h) {
        float v = sp[h] + tp[h];
        v = v > 0.f ? v : NEG_SLOPE * v;
        float ee = __expf(v - __int_as_float(mp[h]));
        eexp[(e << 2) + h] = ee;
        atomAddF32(qp + h, ee);
    }
}

// ---- pass 3: alpha = exp / (sum + eps), in place ---------------------------
__global__ void alpha_kernel(const int* __restrict__ ei, const int* __restrict__ et,
                             const float* __restrict__ esum, float* __restrict__ alpha) {
    int t = blockIdx.x * blockDim.x + threadIdx.x;     // E*H
    if (t >= NED * NH) return;
    int e = t >> 2, h = t & 3;
    int d = ei[NED + e], r = et[e];
    alpha[t] = alpha[t] / (esum[((size_t)r * NND + d) * NH + h] + 1e-8f);
}

// ---- pass 4: out[dst][c] += alpha * H[r][src][c] (HW f32 atomics) ----------
__global__ void scatter_kernel(const int* __restrict__ ei, const int* __restrict__ et,
                               const float* __restrict__ alpha, const float* __restrict__ H,
                               float* __restrict__ out) {
    int t = blockIdx.x * blockDim.x + threadIdx.x;     // E*128 = 102.4M, fits int
    if (t >= NED * HD) return;
    int e = t >> 7, c = t & 127, h = c >> 5;
    int s = ei[e], d = ei[NED + e], r = et[e];
    float a = alpha[(e << 2) + h];
    float val = a * H[((size_t)r * NND + s) * HD + c];
    atomAddF32(out + (size_t)d * HD + c, val);
}

// ---- bias + residual + LayerNorm, one wave32 per node ----------------------
__global__ void finalize_kernel(const float* __restrict__ x, const float* __restrict__ bias,
                                const float* __restrict__ gamma, const float* __restrict__ beta,
                                float* __restrict__ out) {
    int wave = threadIdx.x >> 5, lane = threadIdx.x & 31;
    int n = blockIdx.x * 8 + wave;
    if (n >= NND) return;
    float4 v  = ((const float4*)out)[(size_t)n * 32 + lane];
    float4 xv = ((const float4*)x)[(size_t)n * 32 + lane];
    float4 bv = ((const float4*)bias)[lane];
    v.x += xv.x + bv.x; v.y += xv.y + bv.y; v.z += xv.z + bv.z; v.w += xv.w + bv.w;
    float sum = v.x + v.y + v.z + v.w;
    #pragma unroll
    for (int off = 16; off > 0; off >>= 1) sum += __shfl_xor(sum, off, 32);
    float mu = sum * (1.f / 128.f);
    float dx = v.x - mu, dy = v.y - mu, dz = v.z - mu, dw = v.w - mu;
    float ss = dx * dx + dy * dy + dz * dz + dw * dw;
    #pragma unroll
    for (int off = 16; off > 0; off >>= 1) ss += __shfl_xor(ss, off, 32);
    float rs = rsqrtf(ss * (1.f / 128.f) + LN_EPS);
    float4 g  = ((const float4*)gamma)[lane];
    float4 be = ((const float4*)beta)[lane];
    float4 o;
    o.x = dx * rs * g.x + be.x; o.y = dy * rs * g.y + be.y;
    o.z = dz * rs * g.z + be.z; o.w = dw * rs * g.w + be.w;
    ((float4*)out)[(size_t)n * 32 + lane] = o;
}

extern "C" void kernel_launch(void* const* d_in, const int* in_sizes, int n_in,
                              void* d_out, int out_size, void* d_ws, size_t ws_size,
                              hipStream_t stream) {
    const float* x     = (const float*)d_in[0];
    const int*   ei    = (const int*)d_in[1];
    const int*   et    = (const int*)d_in[2];
    const float* W     = (const float*)d_in[3];
    const float* asrc  = (const float*)d_in[4];
    const float* adst  = (const float*)d_in[5];
    const float* bias  = (const float*)d_in[6];
    const float* gamma = (const float*)d_in[7];
    const float* beta  = (const float*)d_in[8];
    float* out = (float*)d_out;

    char* p = (char*)d_ws;
    float*    Hbuf = (float*)p;    p += (size_t)NR * NND * HD * 4;   // 102.4 MB
    unsigned* xb   = (unsigned*)p; p += (size_t)NND * 64 * 4;        // 12.8 MB
    unsigned* wt   = (unsigned*)p; p += (size_t)NR * HD * 64 * 4;    // 128 KB
    float*    sval = (float*)p;    p += (size_t)NR * NND * NH * 4;   // 3.2 MB
    float*    tval = (float*)p;    p += (size_t)NR * NND * NH * 4;
    int*      emax = (int*)p;      p += (size_t)NR * NND * NH * 4;
    float*    esum = (float*)p;    p += (size_t)NR * NND * NH * 4;
    float*    eexp = (float*)p;    p += (size_t)NED * NH * 4;        // 12.8 MB

    hipMemsetAsync(emax, 0, (size_t)NR * NND * NH * 4, stream);
    hipMemsetAsync(esum, 0, (size_t)NR * NND * NH * 4, stream);
    hipMemsetAsync(out,  0, (size_t)NND * HD * 4, stream);

    cvt_x_kernel<<<(NND * 64 + 255) / 256, 256, 0, stream>>>(x, xb, NND * 64);
    cvt_w_kernel<<<(NR * HD * 64 + 255) / 256, 256, 0, stream>>>(W, wt);
    gemm_kernel<<<dim3((NND + 127) / 128, NR), 256, 0, stream>>>(xb, wt, Hbuf);
    node_attn_kernel<<<(NND * NR * NH + 255) / 256, 256, 0, stream>>>(Hbuf, asrc, adst, sval, tval);
    edge_max_kernel<<<(NED + 255) / 256, 256, 0, stream>>>(ei, et, sval, tval, emax);
    edge_sum_kernel<<<(NED + 255) / 256, 256, 0, stream>>>(ei, et, sval, tval, emax, esum, eexp);
    alpha_kernel<<<(NED * NH + 255) / 256, 256, 0, stream>>>(ei, et, esum, eexp);
    scatter_kernel<<<(NED * HD + 255) / 256, 256, 0, stream>>>(ei, et, eexp, Hbuf, out);
    finalize_kernel<<<(NND + 7) / 8, 256, 0, stream>>>(x, bias, gamma, beta, out);
}
